// HierarchicalCooccurrenceTransformer_37074157699302
// MI455X (gfx1250) — compile-verified, matrix operand order checked by reading, not run on hardware
//
#include <hip/hip_runtime.h>
#include <hip/hip_bf16.h>
#include <math.h>

// ---------------------------------------------------------------------------
// Problem constants (from reference)
// ---------------------------------------------------------------------------
#define BDIM   32
#define WDIM   64
#define LDIM   32
#define DDIM   256
#define HEADS  8
#define HD     32
#define DFFDIM 1024
#define M_INTRA (BDIM * WDIM * LDIM)   // 65536 tokens
#define M_INTER (BDIM * WDIM)          // 2048 tokens
#define QKVN   (3 * DDIM)              // 768 fused QKV width

// ---------------------------------------------------------------------------
// WMMA fragment types (wave32, v_wmma_f32_16x16x32_bf16)
// ---------------------------------------------------------------------------
typedef __attribute__((ext_vector_type(16))) __bf16 v16bf;
typedef __attribute__((ext_vector_type(8)))  float  v8f;
typedef __attribute__((ext_vector_type(4)))  unsigned int u32x4;
typedef __attribute__((ext_vector_type(8)))  int   i32x8;
typedef __attribute__((ext_vector_type(4)))  int   i32x4;

union FragBf { v16bf v; unsigned short u[16]; };

__device__ __forceinline__ unsigned short f2bf(float f) {
  unsigned int u = __float_as_uint(f);
  unsigned int r = (u + 0x7FFFu + ((u >> 16) & 1u)) >> 16;  // RNE
  return (unsigned short)r;
}

// A-matrix (16x32 MxK bf16): lanes 0-15 -> M=lane; lanes 16-31 -> M=lane-16.
// Per-lane vgpr i<4 -> K base 0, i>=4 -> K base 16; +8 for upper lane half.
__device__ __forceinline__ void load_fragA(FragBf& A, const unsigned short* T,
                                           int ld, int mo, int ko, int lane) {
  const int half = lane >> 4, m = lane & 15;
#pragma unroll
  for (int i = 0; i < 8; ++i) {
#pragma unroll
    for (int e = 0; e < 2; ++e) {
      const int k = ((i < 4) ? 0 : 16) + half * 8 + (i & 3) * 2 + e;
      A.u[2 * i + e] = T[(mo + m) * ld + ko + k];
    }
  }
}

// B-matrix (32x16 KxN bf16), row-major source: B[k][n] = T[(ko+k)*ld + no+n]
__device__ __forceinline__ void load_fragB_rm(FragBf& B, const unsigned short* T,
                                              int ld, int ko, int no, int lane) {
  const int half = lane >> 4, n = lane & 15;
#pragma unroll
  for (int i = 0; i < 8; ++i) {
#pragma unroll
    for (int e = 0; e < 2; ++e) {
      const int k = ((i < 4) ? 0 : 16) + half * 8 + (i & 3) * 2 + e;
      B.u[2 * i + e] = T[(ko + k) * ld + no + n];
    }
  }
}

// B-matrix from a transposed source (for Q@K^T): B[k][n] = T[(no+n)*ld + ko+k]
__device__ __forceinline__ void load_fragB_tr(FragBf& B, const unsigned short* T,
                                              int ld, int ko, int no, int lane) {
  const int half = lane >> 4, n = lane & 15;
#pragma unroll
  for (int i = 0; i < 8; ++i) {
#pragma unroll
    for (int e = 0; e < 2; ++e) {
      const int k = ((i < 4) ? 0 : 16) + half * 8 + (i & 3) * 2 + e;
      B.u[2 * i + e] = T[(no + n) * ld + ko + k];
    }
  }
}

// C/D fragment (16x16 f32): M = r + 8*(lane>=16), N = lane&15
__device__ __forceinline__ void load_fragC(v8f& C, const float* T, int ld,
                                           int mo, int no, int lane) {
  const int half = lane >> 4, n = lane & 15;
#pragma unroll
  for (int r = 0; r < 8; ++r) C[r] = T[(mo + r + 8 * half) * ld + no + n];
}

__device__ __forceinline__ v8f wmma_bf16(const FragBf& A, const FragBf& B, v8f C) {
  return __builtin_amdgcn_wmma_f32_16x16x32_bf16(false, A.v, false, B.v,
                                                 (short)0, C, false, false);
}

// ---------------------------------------------------------------------------
// Tensor Data Mover: 2-D bf16 tile (tile_w elems x tile_h rows, row stride in
// elems) from global into LDS at byte offset lds_off (workgroup LDS base).
// D# layout per CDNA5 ISA.
// This toolchain ships the 6-arg builtin:
//   (uint32x4 g0, int32x8 g1, int32x4 g2, int32x4 g3, int32x8 pad, i32 cpol)
// ---------------------------------------------------------------------------
#if defined(__AMDGCN__) && __has_builtin(__builtin_amdgcn_tensor_load_to_lds)
#define HAVE_TDM 1
#else
#define HAVE_TDM 0
#endif

#if HAVE_TDM
__device__ __forceinline__ void tdm_load_2d_bf16(unsigned lds_off, const void* gptr,
                                                 int tile_w, int tile_h,
                                                 int row_stride) {
  const unsigned long long ga = (unsigned long long)(uintptr_t)gptr;
  u32x4 g0;
  g0[0] = 1u;                                               // count=1, user D#
  g0[1] = lds_off;                                          // lds_addr (bytes)
  g0[2] = (unsigned)(ga & 0xFFFFFFFFull);                   // global_addr lo
  g0[3] = (unsigned)((ga >> 32) & 0x01FFFFFFull) | (2u << 30);  // addr hi | type=2
  i32x8 g1;
  g1[0] = 1 << 16;                                          // data_size=2 bytes
  g1[1] = (int)(((unsigned)tile_w & 0xFFFFu) << 16);        // tensor_dim0 lo16
  g1[2] = (int)(((unsigned)tile_w >> 16) |
                (((unsigned)tile_h & 0xFFFFu) << 16));      // dim0 hi | dim1 lo
  g1[3] = (int)(((unsigned)tile_h >> 16) |
                ((unsigned)tile_w << 16));                  // dim1 hi | tile_dim0
  g1[4] = tile_h;                                           // tile_dim1 | tile_dim2=0
  g1[5] = row_stride;                                       // tensor_dim0_stride lo32
  g1[6] = 0;                                                // stride0 hi | stride1 lo
  g1[7] = 0;
  i32x4 gz;
  gz[0] = 0; gz[1] = 0; gz[2] = 0; gz[3] = 0;
  i32x8 gz8;
  gz8[0] = 0; gz8[1] = 0; gz8[2] = 0; gz8[3] = 0;
  gz8[4] = 0; gz8[5] = 0; gz8[6] = 0; gz8[7] = 0;
  __builtin_amdgcn_tensor_load_to_lds(g0, g1, gz, gz, gz8, 0);
}
#endif

// ---------------------------------------------------------------------------
// f32 -> bf16 conversion (weights)
// ---------------------------------------------------------------------------
__global__ void cvt_bf16_kernel(const float* __restrict__ src,
                                unsigned short* __restrict__ dst, int n) {
  int i = blockIdx.x * 256 + threadIdx.x;
  if (i < n) dst[i] = f2bf(src[i]);
}

// pack wq|wk|wv -> [256, 768] bf16
__global__ void pack_qkv_w_kernel(const float* __restrict__ wq,
                                  const float* __restrict__ wk,
                                  const float* __restrict__ wv,
                                  unsigned short* __restrict__ dst) {
  const int idx = blockIdx.x * 256 + threadIdx.x;   // over 256*768
  const int k = idx / QKVN, c = idx % QKVN;
  const float* src = (c < DDIM) ? wq : (c < 2 * DDIM) ? wk : wv;
  dst[idx] = f2bf(src[k * DDIM + (c & (DDIM - 1))]);
}

__global__ void pack_qkv_b_kernel(const float* __restrict__ bq,
                                  const float* __restrict__ bk,
                                  const float* __restrict__ bv,
                                  float* __restrict__ dst) {
  const int c = blockIdx.x * 256 + threadIdx.x;     // over 768
  dst[c] = (c < DDIM) ? bq[c] : (c < 2 * DDIM) ? bk[c - DDIM] : bv[c - 2 * DDIM];
}

// ---------------------------------------------------------------------------
// Embedding: x = node_emb + type_emb[0] + sinusoidal_pe(l, d); write f32 + bf16
// ---------------------------------------------------------------------------
__global__ void embed_kernel(const float* __restrict__ ne,
                             const float* __restrict__ type_emb,
                             float* __restrict__ x,
                             unsigned short* __restrict__ xbf) {
  const int tok = blockIdx.x;
  const int d   = threadIdx.x;
  const int l   = tok & (LDIM - 1);
  const size_t idx = (size_t)tok * DDIM + d;
  const float freq = __expf((float)(d & ~1) * (-9.210340371976184f / (float)DDIM));
  const float arg  = (float)l * freq;
  const float pe   = (d & 1) ? __cosf(arg) : __sinf(arg);
  const float v = ne[idx] + type_emb[d] + pe;   // walk_type = 0 row
  x[idx]   = v;
  xbf[idx] = f2bf(v);
}

// ---------------------------------------------------------------------------
// Cooccurrence: one wave per (b,r,s). lane = column j over L=32.
// ---------------------------------------------------------------------------
__global__ __launch_bounds__(32)
void cooccur_kernel(const int* __restrict__ an, const int* __restrict__ wm,
                    float* __restrict__ co) {
  const int idx = blockIdx.x;
  const int s = idx & (WDIM - 1);
  const int r = (idx >> 6) & (WDIM - 1);
  const int b = idx >> 12;
  const int lane = threadIdx.x;

  const int* ar = an + ((size_t)(b * WDIM + r)) * LDIM;
  const int* as = an + ((size_t)(b * WDIM + s)) * LDIM;
  const int* mr = wm + ((size_t)(b * WDIM + r)) * LDIM;
  const int* ms = wm + ((size_t)(b * WDIM + s)) * LDIM;

  const int aj = as[lane];
  const int mj = ms[lane];
  float acc = 0.f;
  int vr = 0;
#pragma unroll
  for (int i = 0; i < LDIM; ++i) {
    const int ai = ar[i];
    const int mi_ = mr[i];
    vr += (mi_ != 0);
    if (mj && mi_ && (ai == aj)) {
      const float dd = (float)(i - lane);
      acc += __expf(-dd * dd * 0.25f);   // sigma^2 = 4
    }
  }
  int vs_ = (mj != 0);
#pragma unroll
  for (int off = 16; off > 0; off >>= 1) {
    acc += __shfl_xor(acc, off, 32);
    vs_ += __shfl_xor(vs_, off, 32);
  }
  if (lane == 0) co[idx] = acc / ((float)vr * (float)vs_ + 1e-8f);
}

__global__ void walk_valid_kernel(const int* __restrict__ wm,
                                  int* __restrict__ wv, int n) {
  int i = blockIdx.x * blockDim.x + threadIdx.x;
  if (i >= n) return;
  int s = 0;
#pragma unroll
  for (int l = 0; l < LDIM; ++l) s += wm[(size_t)i * LDIM + l];
  wv[i] = (s > 0) ? 1 : 0;
}

// ---------------------------------------------------------------------------
// Tiled bf16 WMMA GEMM:  Y[M,N] = X[M,K] @ W[K,N] + bias  (optional ReLU)
// block = 128 threads (4 waves), block tile 64x64, each wave owns 16x64.
// Weight tile is fetched by the Tensor Data Mover (wave 0), X tile by vector
// loads from all threads (overlapped).
// ---------------------------------------------------------------------------
#define GEMM_XS_BYTES (64 * 32 * 2)
#define GEMM_WS_OFF   GEMM_XS_BYTES

template <int RELU>
__global__ __launch_bounds__(128)
void gemm_bf16_kernel(const unsigned short* __restrict__ X,
                      const unsigned short* __restrict__ W,
                      const float* __restrict__ bias,
                      float* __restrict__ Yf,
                      unsigned short* __restrict__ Ybf,
                      int M, int N, int K) {
  __shared__ alignas(16) unsigned short smem[64 * 32 + 32 * 64];
  unsigned short* Xs = smem;
  unsigned short* Ws = smem + 64 * 32;

  const int tid  = threadIdx.x;
  const int lane = tid & 31;
  const int wid  = tid >> 5;
  const int gm0  = blockIdx.y * 64;
  const int gn0  = blockIdx.x * 64;

  v8f acc[4] = {v8f{}, v8f{}, v8f{}, v8f{}};

  for (int kk = 0; kk < K; kk += 32) {
    // X tile 64x32: 256 uint4 chunks over 128 threads
#pragma unroll
    for (int t = 0; t < 2; ++t) {
      const int ch = tid + t * 128;
      const int r = ch >> 2, cc = ch & 3;
      *(uint4*)(Xs + r * 32 + cc * 8) =
          *(const uint4*)(X + (size_t)(gm0 + r) * K + kk + cc * 8);
    }
#if HAVE_TDM
    if (wid == 0) {
      tdm_load_2d_bf16(GEMM_WS_OFF, W + (size_t)kk * N + gn0, 64, 32, N);
      __builtin_amdgcn_s_wait_tensorcnt(0);
    }
#else
#pragma unroll
    for (int t = 0; t < 2; ++t) {
      const int ch = tid + t * 128;
      const int r = ch >> 3, cc = ch & 7;
      *(uint4*)(Ws + r * 64 + cc * 8) =
          *(const uint4*)(W + (size_t)(kk + r) * N + gn0 + cc * 8);
    }
#endif
    __syncthreads();

    FragBf A;
    load_fragA(A, Xs, 32, wid * 16, 0, lane);
#pragma unroll
    for (int ni = 0; ni < 4; ++ni) {
      FragBf Bf;
      load_fragB_rm(Bf, Ws, 64, 0, ni * 16, lane);
      acc[ni] = wmma_bf16(A, Bf, acc[ni]);
    }
    __syncthreads();
  }

  const int mofs = wid * 16 + ((lane & 16) ? 8 : 0);
#pragma unroll
  for (int ni = 0; ni < 4; ++ni) {
#pragma unroll
    for (int r = 0; r < 8; ++r) {
      const int row = gm0 + mofs + r;
      const int col = gn0 + ni * 16 + (lane & 15);
      float o = acc[ni][r] + bias[col];
      if (RELU) o = fmaxf(o, 0.f);
      const size_t oi = (size_t)row * N + col;
      if (Yf)  Yf[oi]  = o;
      if (Ybf) Ybf[oi] = f2bf(o);
    }
  }
}

// ---------------------------------------------------------------------------
// Attention: one wave per (sequence n, head h). hd = 32, so each 16x16 score
// tile is a single K=32 WMMA. QKV is the fused [M, 768] buffer (row stride rs).
// Bias (+key mask) is pre-scaled and fed as the WMMA C operand.
// ---------------------------------------------------------------------------
template <int S>
__global__ __launch_bounds__(32)
void attention_kernel(const unsigned short* __restrict__ QKV, int rs,
                      unsigned short* __restrict__ O,
                      const int* __restrict__ kvalid,
                      const float* __restrict__ co,
                      const float* __restrict__ gamma,
                      int Wd) {
  const int blk  = blockIdx.x;
  const int h    = blk & (HEADS - 1);
  const int n    = blk >> 3;
  const int lane = threadIdx.x;

  constexpr int QKV_BYTES = 3 * S * HD * 2;
  constexpr int SC_BYTES  = S * (S + 1) * 4;
  constexpr int AS_BYTES  = S * S * 2;
  __shared__ alignas(16) unsigned char smem[QKV_BYTES + SC_BYTES + AS_BYTES];
  unsigned short* qs  = (unsigned short*)smem;
  unsigned short* ks  = qs + S * HD;
  unsigned short* vs  = ks + S * HD;
  float*          sc  = (float*)(smem + QKV_BYTES);
  unsigned short* as_ = (unsigned short*)(smem + QKV_BYTES + SC_BYTES);

  const size_t rowbase = (size_t)(n * S) * rs + h * HD;
#if HAVE_TDM
  // Q/K/V head tiles are strided 2-D tiles: S rows x 32 bf16, row stride rs.
  tdm_load_2d_bf16(0,              QKV + rowbase,            HD, S, rs);
  tdm_load_2d_bf16(S * HD * 2,     QKV + rowbase + DDIM,     HD, S, rs);
  tdm_load_2d_bf16(2 * S * HD * 2, QKV + rowbase + 2 * DDIM, HD, S, rs);
  __builtin_amdgcn_s_wait_tensorcnt(0);
#else
  for (int s = lane; s < S; s += 32) {
    const size_t base = rowbase + (size_t)s * rs;
#pragma unroll
    for (int c = 0; c < 4; ++c) {
      ((uint4*)(qs + s * HD))[c] = ((const uint4*)(QKV + base))[c];
      ((uint4*)(ks + s * HD))[c] = ((const uint4*)(QKV + base + DDIM))[c];
      ((uint4*)(vs + s * HD))[c] = ((const uint4*)(QKV + base + 2 * DDIM))[c];
    }
  }
#endif

  const float scale     = 0.17677669529663687f;   // 1/sqrt(32)
  const float inv_scale = 5.656854249492381f;     // sqrt(32)
  const float gco = (co != nullptr) ? gamma[0] : 0.f;
  constexpr int MT = S / 16;

  // Pre-fill score LDS with (bias + key mask) / scale -> becomes the C operand.
  for (int t = lane; t < S * S; t += 32) {
    const int mrow = t / S, col = t - mrow * S;
    float bv = (kvalid[n * S + col] == 0) ? -1e9f : 0.f;
    if (co) bv += gco * co[(size_t)n * Wd * Wd + (size_t)mrow * Wd + col];
    sc[mrow * (S + 1) + col] = bv * inv_scale;
  }
  __syncthreads();

  // scores = scale * (Q@K^T + C)
#pragma unroll
  for (int mi = 0; mi < MT; ++mi) {
    FragBf A;
    load_fragA(A, qs, HD, mi * 16, 0, lane);
#pragma unroll
    for (int ni = 0; ni < MT; ++ni) {
      FragBf Bf;
      load_fragB_tr(Bf, ks, HD, 0, ni * 16, lane);
      v8f cfr;
      load_fragC(cfr, sc, S + 1, mi * 16, ni * 16, lane);
      v8f accv = wmma_bf16(A, Bf, cfr);
#pragma unroll
      for (int r = 0; r < 8; ++r) {
        const int mrow = mi * 16 + r + ((lane & 16) ? 8 : 0);
        const int col  = ni * 16 + (lane & 15);
        sc[mrow * (S + 1) + col] = accv[r] * scale;
      }
    }
  }
  __syncthreads();

  // row softmax -> bf16 probabilities
  for (int row = lane; row < S; row += 32) {
    float mx = -1e30f;
#pragma unroll
    for (int j = 0; j < S; ++j) mx = fmaxf(mx, sc[row * (S + 1) + j]);
    float sum = 0.f;
#pragma unroll
    for (int j = 0; j < S; ++j) {
      const float e = __expf(sc[row * (S + 1) + j] - mx);
      sc[row * (S + 1) + j] = e;
      sum += e;
    }
    const float inv = 1.f / sum;
#pragma unroll
    for (int j = 0; j < S; ++j) as_[row * S + j] = f2bf(sc[row * (S + 1) + j] * inv);
  }
  __syncthreads();

  // out = P @ V  (P: SxS bf16, V: Sx32)
#pragma unroll
  for (int mi = 0; mi < MT; ++mi) {
    v8f acc0 = {}, acc1 = {};
#pragma unroll
    for (int kk = 0; kk < S; kk += 32) {
      FragBf A;
      load_fragA(A, as_, S, mi * 16, kk, lane);
      FragBf B0, B1;
      load_fragB_rm(B0, vs, HD, kk, 0, lane);
      acc0 = wmma_bf16(A, B0, acc0);
      load_fragB_rm(B1, vs, HD, kk, 16, lane);
      acc1 = wmma_bf16(A, B1, acc1);
    }
#pragma unroll
    for (int r = 0; r < 8; ++r) {
      const int srow = mi * 16 + r + ((lane & 16) ? 8 : 0);
      const int col  = lane & 15;
      const size_t base = ((size_t)(n * S + srow)) * DDIM + h * HD;
      O[base + col]      = f2bf(acc0[r]);
      O[base + 16 + col] = f2bf(acc1[r]);
    }
  }
}

// ---------------------------------------------------------------------------
// Residual + LayerNorm (f==nullptr -> plain LN). One wave per row of 256.
// ---------------------------------------------------------------------------
__global__ __launch_bounds__(32)
void residual_ln_kernel(const float* __restrict__ x, const float* __restrict__ f,
                        const float* __restrict__ g, const float* __restrict__ b,
                        float* __restrict__ yout, unsigned short* __restrict__ ybf) {
  const int row  = blockIdx.x;
  const int lane = threadIdx.x;
  float v[8];
  const size_t base = (size_t)row * DDIM;
#pragma unroll
  for (int i = 0; i < 8; ++i) {
    const int d = lane + i * 32;
    v[i] = x[base + d];
    if (f) v[i] += f[base + d];
  }
  float s = 0.f;
#pragma unroll
  for (int i = 0; i < 8; ++i) s += v[i];
#pragma unroll
  for (int off = 16; off > 0; off >>= 1) s += __shfl_xor(s, off, 32);
  const float mu = s * (1.f / DDIM);
  float q = 0.f;
#pragma unroll
  for (int i = 0; i < 8; ++i) { const float d = v[i] - mu; q += d * d; }
#pragma unroll
  for (int off = 16; off > 0; off >>= 1) q += __shfl_xor(q, off, 32);
  const float rs = rsqrtf(q * (1.f / DDIM) + 1e-5f);
#pragma unroll
  for (int i = 0; i < 8; ++i) {
    const int d = lane + i * 32;
    const float o = (v[i] - mu) * rs * g[d] + b[d];
    yout[base + d] = o;
    if (ybf) ybf[base + d] = f2bf(o);
  }
}

// ---------------------------------------------------------------------------
// Masked mean pooling over L tokens -> pooled bf16 [B*W, 256]
// ---------------------------------------------------------------------------
__global__ __launch_bounds__(32)
void pool_kernel(const float* __restrict__ x, const int* __restrict__ wm,
                 unsigned short* __restrict__ pooled) {
  const int n = blockIdx.x;
  const int lane = threadIdx.x;
  float acc[8] = {0, 0, 0, 0, 0, 0, 0, 0};
  float cnt = 0.f;
  for (int l = 0; l < LDIM; ++l) {
    const int m = wm[(size_t)n * LDIM + l];
    cnt += (float)m;
    if (m) {
      const float* row = x + ((size_t)(n * LDIM + l)) * DDIM;
#pragma unroll
      for (int i = 0; i < 8; ++i) acc[i] += row[lane + i * 32];
    }
  }
  const float inv = 1.f / (cnt + 1e-8f);
#pragma unroll
  for (int i = 0; i < 8; ++i)
    pooled[(size_t)n * DDIM + lane + i * 32] = f2bf(acc[i] * inv);
}

// ---------------------------------------------------------------------------
// Host-side orchestration.
// d_in: 0 node_emb  1 anon  2 masks  3 gamma
//   4..19 inter_layers[0] {b1,b2,bk,bo,bq,bv,ln1_b,ln1_g,ln2_b,ln2_g,w1,w2,wk,wo,wq,wv}
//  20..35 inter_layers[1]  36 inter_norm_b  37 inter_norm_g
//  38..53 intra_layers[0]  54..69 intra_layers[1]
//  70 intra_norm_b  71 intra_norm_g  72 proj_b  73 proj_w  74 type_emb
// ---------------------------------------------------------------------------
struct LayerP {
  const float *b1, *b2, *bk, *bo, *bq, *bv;
  const float *ln1b, *ln1g, *ln2b, *ln2g;
  const float *w1, *w2, *wk, *wo, *wq, *wv;
  unsigned short *wqkv_bf, *wo_bf, *w1_bf, *w2_bf;
  float *bqkv;
};

static LayerP make_layer(void* const* d_in, int base) {
  LayerP L;
  L.b1 = (const float*)d_in[base + 0];  L.b2 = (const float*)d_in[base + 1];
  L.bk = (const float*)d_in[base + 2];  L.bo = (const float*)d_in[base + 3];
  L.bq = (const float*)d_in[base + 4];  L.bv = (const float*)d_in[base + 5];
  L.ln1b = (const float*)d_in[base + 6]; L.ln1g = (const float*)d_in[base + 7];
  L.ln2b = (const float*)d_in[base + 8]; L.ln2g = (const float*)d_in[base + 9];
  L.w1 = (const float*)d_in[base + 10]; L.w2 = (const float*)d_in[base + 11];
  L.wk = (const float*)d_in[base + 12]; L.wo = (const float*)d_in[base + 13];
  L.wq = (const float*)d_in[base + 14]; L.wv = (const float*)d_in[base + 15];
  return L;
}

template <int S>
static void run_layer(hipStream_t stream, const LayerP& L,
                      float* x, unsigned short* xbf, unsigned short* qkvbf,
                      unsigned short* abf, float* of32, unsigned short* hbf,
                      int M, const int* kvalid, const float* co,
                      const float* gamma, int Wd) {
  const int Nseq = M / S;
  // fused QKV projection (bf16 output only)
  gemm_bf16_kernel<0><<<dim3(QKVN / 64, M / 64), 128, 0, stream>>>(
      xbf, L.wqkv_bf, L.bqkv, nullptr, qkvbf, M, QKVN, DDIM);
  // attention
  attention_kernel<S><<<Nseq * HEADS, 32, 0, stream>>>(
      qkvbf, QKVN, abf, kvalid, co, gamma, Wd);
  // output projection (f32)
  gemm_bf16_kernel<0><<<dim3(DDIM / 64, M / 64), 128, 0, stream>>>(
      abf, L.wo_bf, L.bo, of32, nullptr, M, DDIM, DDIM);
  // x = LN(x + attn)
  residual_ln_kernel<<<M, 32, 0, stream>>>(x, of32, L.ln1g, L.ln1b, x, xbf);
  // FFN: relu(x@w1+b1)@w2+b2
  gemm_bf16_kernel<1><<<dim3(DFFDIM / 64, M / 64), 128, 0, stream>>>(
      xbf, L.w1_bf, L.b1, nullptr, hbf, M, DFFDIM, DDIM);
  gemm_bf16_kernel<0><<<dim3(DDIM / 64, M / 64), 128, 0, stream>>>(
      hbf, L.w2_bf, L.b2, of32, nullptr, M, DDIM, DFFDIM);
  // x = LN(x + ffn)
  residual_ln_kernel<<<M, 32, 0, stream>>>(x, of32, L.ln2g, L.ln2b, x, xbf);
}

extern "C" void kernel_launch(void* const* d_in, const int* in_sizes, int n_in,
                              void* d_out, int out_size, void* d_ws, size_t ws_size,
                              hipStream_t stream) {
  (void)in_sizes; (void)n_in; (void)out_size; (void)ws_size;

  const float* node_emb = (const float*)d_in[0];
  const int*   anon     = (const int*)d_in[1];
  const int*   masks    = (const int*)d_in[2];
  const float* gamma    = (const float*)d_in[3];
  LayerP inter0 = make_layer(d_in, 4);
  LayerP inter1 = make_layer(d_in, 20);
  const float* inter_nb = (const float*)d_in[36];
  const float* inter_ng = (const float*)d_in[37];
  LayerP intra0 = make_layer(d_in, 38);
  LayerP intra1 = make_layer(d_in, 54);
  const float* intra_nb = (const float*)d_in[70];
  const float* intra_ng = (const float*)d_in[71];
  const float* proj_b   = (const float*)d_in[72];
  const float* proj_w   = (const float*)d_in[73];
  const float* type_emb = (const float*)d_in[74];

  // ---- workspace bump allocator ----
  char* wsp = (char*)d_ws;
  auto alloc = [&](size_t bytes) -> void* {
    void* p = (void*)wsp;
    wsp += (bytes + 255) & ~(size_t)255;
    return p;
  };
  float*          x     = (float*)alloc((size_t)M_INTRA * DDIM * 4);
  unsigned short* xbf   = (unsigned short*)alloc((size_t)M_INTRA * DDIM * 2);
  unsigned short* qkvbf = (unsigned short*)alloc((size_t)M_INTRA * QKVN * 2);
  unsigned short* abf   = (unsigned short*)alloc((size_t)M_INTRA * DDIM * 2);
  float*          of32  = (float*)alloc((size_t)M_INTRA * DDIM * 4);
  unsigned short* hbf   = (unsigned short*)alloc((size_t)M_INTRA * DFFDIM * 2);
  float*          co    = (float*)alloc((size_t)BDIM * WDIM * WDIM * 4);
  int*            wval  = (int*)alloc((size_t)M_INTER * 4);
  unsigned short* pbf   = (unsigned short*)alloc((size_t)M_INTER * DDIM * 2);
  float*          y     = (float*)alloc((size_t)M_INTER * DDIM * 4);
  unsigned short* ybf   = (unsigned short*)alloc((size_t)M_INTER * DDIM * 2);
  unsigned short* pwbf  = (unsigned short*)alloc((size_t)DDIM * DDIM * 2);

  LayerP* layers[4] = {&intra0, &intra1, &inter0, &inter1};
  for (int i = 0; i < 4; ++i) {
    layers[i]->wqkv_bf = (unsigned short*)alloc((size_t)DDIM * QKVN * 2);
    layers[i]->wo_bf   = (unsigned short*)alloc((size_t)DDIM * DDIM * 2);
    layers[i]->w1_bf   = (unsigned short*)alloc((size_t)DDIM * DFFDIM * 2);
    layers[i]->w2_bf   = (unsigned short*)alloc((size_t)DFFDIM * DDIM * 2);
    layers[i]->bqkv    = (float*)alloc((size_t)QKVN * 4);
  }

  // ---- weight conversion / packing ----
  auto cvt = [&](const float* s, unsigned short* d, int n) {
    cvt_bf16_kernel<<<(n + 255) / 256, 256, 0, stream>>>(s, d, n);
  };
  for (int i = 0; i < 4; ++i) {
    pack_qkv_w_kernel<<<(DDIM * QKVN) / 256, 256, 0, stream>>>(
        layers[i]->wq, layers[i]->wk, layers[i]->wv, layers[i]->wqkv_bf);
    pack_qkv_b_kernel<<<QKVN / 256, 256, 0, stream>>>(
        layers[i]->bq, layers[i]->bk, layers[i]->bv, layers[i]->bqkv);
    cvt(layers[i]->wo, layers[i]->wo_bf, DDIM * DDIM);
    cvt(layers[i]->w1, layers[i]->w1_bf, DDIM * DFFDIM);
    cvt(layers[i]->w2, layers[i]->w2_bf, DFFDIM * DDIM);
  }
  cvt(proj_w, pwbf, DDIM * DDIM);

  // ---- embedding + PE ----
  embed_kernel<<<M_INTRA, DDIM, 0, stream>>>(node_emb, type_emb, x, xbf);

  // ---- cooccurrence + walk validity ----
  cooccur_kernel<<<BDIM * WDIM * WDIM, 32, 0, stream>>>(anon, masks, co);
  walk_valid_kernel<<<(M_INTER + 255) / 256, 256, 0, stream>>>(masks, wval, M_INTER);

  // ---- intra-walk encoder (S = 32, key mask from walk_masks, no bias) ----
  run_layer<LDIM>(stream, intra0, x, xbf, qkvbf, abf, of32, hbf,
                  M_INTRA, masks, nullptr, nullptr, 0);
  run_layer<LDIM>(stream, intra1, x, xbf, qkvbf, abf, of32, hbf,
                  M_INTRA, masks, nullptr, nullptr, 0);
  residual_ln_kernel<<<M_INTRA, 32, 0, stream>>>(x, nullptr, intra_ng, intra_nb,
                                                 x, nullptr);

  // ---- masked mean pooling + projection ----
  pool_kernel<<<M_INTER, 32, 0, stream>>>(x, masks, pbf);
  gemm_bf16_kernel<0><<<dim3(DDIM / 64, M_INTER / 64), 128, 0, stream>>>(
      pbf, pwbf, proj_b, y, ybf, M_INTER, DDIM, DDIM);

  // ---- inter-walk encoder (S = 64, gamma*co bias, walk-level key mask) ----
  run_layer<WDIM>(stream, inter0, y, ybf, qkvbf, abf, of32, hbf,
                  M_INTER, wval, co, gamma, WDIM);
  run_layer<WDIM>(stream, inter1, y, ybf, qkvbf, abf, of32, hbf,
                  M_INTER, wval, co, gamma, WDIM);

  // ---- final norm straight into d_out ----
  residual_ln_kernel<<<M_INTER, 32, 0, stream>>>(y, nullptr, inter_ng, inter_nb,
                                                 (float*)d_out, nullptr);
}
